// MultiAgentTransformer_24610162606163
// MI455X (gfx1250) — matrix operand order repaired and checked by reading, not compiled, not run on hardware
//
#include <hip/hip_runtime.h>
#include <hip/hip_bf16.h>
#include <math.h>

// ---------------------------------------------------------------------------
// MultiAgentTransformer decoder on MI455X (gfx1250, wave32, WMMA).
// Bandwidth-bound: fused bf16-WMMA GEMMs (f32 accumulate), weights
// pre-converted/transposed to bf16 once per call (L2-resident), B tiles
// staged via async global->LDS b128, A tiles via vectorized float4 ->
// packed-bf16 stores. Residuals folded into LayerNorm. QKV fused.
// ---------------------------------------------------------------------------

typedef __bf16  v16bf __attribute__((ext_vector_type(16)));
typedef float   v8f   __attribute__((ext_vector_type(8)));
typedef int     v4i   __attribute__((ext_vector_type(4)));

#define M_ROWS  32768      // B*L = 2048*16
#define L_AG    16
#define HID     20
#define NE      1024

#if defined(__has_builtin)
#if __has_builtin(__builtin_amdgcn_global_load_async_to_lds_b128)
#define HAVE_ASYNC_LDS 1
#endif
#endif
#ifndef HAVE_ASYNC_LDS
#define HAVE_ASYNC_LDS 0
#endif

union Frag { v16bf v; unsigned int u[8]; };

__device__ __forceinline__ unsigned short f2bf(float f) {
    unsigned int u = __float_as_uint(f);
    unsigned int r = u + 0x7FFFu + ((u >> 16) & 1u);   // RNE bf16
    return (unsigned short)(r >> 16);
}

__device__ __forceinline__ float gelu_exact(float x) {
    return 0.5f * x * (1.0f + erff(x * 0.70710678118654752f));
}

// ---------------------------------------------------------------------------
// Weight convert + transpose: Wt[(nOff+n)*K + k] = bf16(W[k*N + n])
// 32x32 tiles through LDS; coalesced fp32 reads, contiguous bf16 writes.
// ---------------------------------------------------------------------------
__global__ __launch_bounds__(256)
void conv_t_kernel(const float* __restrict__ W, unsigned short* __restrict__ Wt,
                   int K, int N, int nOff)
{
    __shared__ unsigned short tile[32][33];
    const int tid = threadIdx.x;
    const int n0 = blockIdx.x * 32;
    const int k0 = blockIdx.y * 32;
    #pragma unroll
    for (int i = 0; i < 4; ++i) {
        int e = tid + i * 256;
        int k = e >> 5, n = e & 31;
        int gk = k0 + k, gn = n0 + n;
        float v = (gk < K && gn < N) ? W[(size_t)gk * N + gn] : 0.0f;
        tile[n][k] = f2bf(v);
    }
    __syncthreads();
    #pragma unroll
    for (int i = 0; i < 4; ++i) {
        int e = tid + i * 256;
        int n = e >> 5, k = e & 31;
        int gn = n0 + n, gk = k0 + k;
        if (gn < N && gk < K)
            Wt[(size_t)(nOff + gn) * K + gk] = tile[n][k];
    }
}

// concat up to 3 bias vectors
__global__ void concat3_kernel(const float* a, const float* b, const float* c,
                               float* out, int na, int nb, int nc)
{
    int t = threadIdx.x;
    if (t < na) out[t] = a[t];
    else if (t < na + nb) out[t] = b[t - na];
    else if (t < na + nb + nc) out[t] = c[t - na - nb];
}

// ---------------------------------------------------------------------------
// GEMM: C[m, n] (ldC) = act(A[M,K] @ Wt^T + bias), Wt is bf16 [N][K].
// 256 threads / 8 waves, 128x64 tile, 32-deep K steps, M % 128 == 0 assumed.
// ---------------------------------------------------------------------------
template<int ACT>
__global__ __launch_bounds__(256)
void gemm_kernel(const float* __restrict__ A, const unsigned short* __restrict__ Wt,
                 const float* __restrict__ bias, float* __restrict__ C,
                 int M, int N, int K, int ldC)
{
    __shared__ unsigned short As[128 * 32];   // [m][k] bf16
    __shared__ unsigned short Bs[64 * 32];    // [n][k] bf16

    const int tid  = threadIdx.x;
    const int m0   = blockIdx.y * 128;
    const int n0   = blockIdx.x * 64;
    const int w    = tid >> 5;
    const int lane = tid & 31;
    const int half = lane >> 4;
    const int l15  = lane & 15;
    const int wm   = w & 3;
    const int wn   = w >> 2;

    const bool a_vec = ((K & 3) == 0);   // float4 rows stay 16B aligned
    const bool b_vec = ((K & 7) == 0);   // bf16 rows stay 16B aligned

    // staging coordinates (computed once)
    const int bn = tid >> 2, bkc = (tid & 3) * 8;     // B: 16B per thread
    const int bgn = n0 + bn;

    v8f acc[2][2] = {};

    for (int kk = 0; kk < K; kk += 32) {
        const bool kfull = (kk + 32 <= K);

        // ---- stage A (128x32): float4 -> packed bf16 pairs ----
        if (a_vec && kfull) {
            #pragma unroll
            for (int i = 0; i < 4; ++i) {
                int e = tid + i * 256;
                int m = e >> 3, c = e & 7;
                const float4 f = *(const float4*)(A + (size_t)(m0 + m) * K + kk + c * 4);
                uint2 p;
                p.x = (unsigned int)f2bf(f.x) | ((unsigned int)f2bf(f.y) << 16);
                p.y = (unsigned int)f2bf(f.z) | ((unsigned int)f2bf(f.w) << 16);
                *(uint2*)&As[m * 32 + c * 4] = p;
            }
        } else {
            #pragma unroll
            for (int i = 0; i < 4; ++i) {
                int e = tid + i * 256;
                int m = e >> 3, c = e & 7;
                float fv[4];
                #pragma unroll
                for (int j = 0; j < 4; ++j) {
                    int gk = kk + c * 4 + j;
                    fv[j] = (gk < K) ? A[(size_t)(m0 + m) * K + gk] : 0.0f;
                }
                uint2 p;
                p.x = (unsigned int)f2bf(fv[0]) | ((unsigned int)f2bf(fv[1]) << 16);
                p.y = (unsigned int)f2bf(fv[2]) | ((unsigned int)f2bf(fv[3]) << 16);
                *(uint2*)&As[m * 32 + c * 4] = p;
            }
        }

        // ---- stage B (64x32): straight bf16 copy, async when possible ----
        if (b_vec && kfull && bgn < N) {
            const unsigned short* src = Wt + (size_t)bgn * K + kk + bkc;
#if HAVE_ASYNC_LDS
            __builtin_amdgcn_global_load_async_to_lds_b128(
                (__attribute__((address_space(1))) v4i*)src,
                (__attribute__((address_space(3))) v4i*)&Bs[bn * 32 + bkc],
                0, 0);
#else
            *(uint4*)&Bs[bn * 32 + bkc] = *(const uint4*)src;
#endif
        } else {
            #pragma unroll
            for (int j = 0; j < 8; ++j) {
                int gk = kk + bkc + j;
                unsigned short v = (bgn < N && gk < K)
                                 ? Wt[(size_t)bgn * K + gk] : (unsigned short)0;
                Bs[bn * 32 + bkc + j] = v;
            }
        }

        // prefetch next A K-slab into L2 (global_prefetch_b8)
        if (kk + 32 < K) {
            __builtin_prefetch(&A[(size_t)(m0 + (tid >> 1)) * K + kk + 32], 0, 0);
        }

#if HAVE_ASYNC_LDS
#if __has_builtin(__builtin_amdgcn_s_wait_asynccnt)
        __builtin_amdgcn_s_wait_asynccnt(0);
#else
        asm volatile("s_wait_asynccnt 0x0" ::: "memory");
#endif
#endif
        __syncthreads();

        // ---- fragments (ISA VGPR layouts) + WMMA ----
        Frag a[2], b[2];
        #pragma unroll
        for (int ti = 0; ti < 2; ++ti)
            #pragma unroll
            for (int r = 0; r < 8; ++r) {
                int k = ((r >> 2) << 4) + (half << 3) + ((r & 3) << 1);
                a[ti].u[r] = *(const unsigned int*)&As[(wm * 32 + ti * 16 + l15) * 32 + k];
            }
        #pragma unroll
        for (int tj = 0; tj < 2; ++tj)
            #pragma unroll
            for (int r = 0; r < 8; ++r) {
                int k = ((r >> 2) << 4) + (half << 3) + ((r & 3) << 1);
                b[tj].u[r] = *(const unsigned int*)&Bs[(wn * 32 + tj * 16 + l15) * 32 + k];
            }
        #pragma unroll
        for (int ti = 0; ti < 2; ++ti)
            #pragma unroll
            for (int tj = 0; tj < 2; ++tj)
                acc[ti][tj] = __builtin_amdgcn_wmma_f32_16x16x32_bf16(
                    false, a[ti].v, false, b[tj].v, (short)0, acc[ti][tj], false, false);
        __syncthreads();
    }

    // ---- epilogue (M % 128 == 0: only N guard, hoisted per column) ----
    #pragma unroll
    for (int tj = 0; tj < 2; ++tj) {
        const int n = n0 + wn * 32 + tj * 16 + l15;
        const bool nok = n < N;
        const float bv = (nok && bias) ? bias[n] : 0.0f;
        #pragma unroll
        for (int ti = 0; ti < 2; ++ti) {
            const int mbase = m0 + wm * 32 + ti * 16 + (half << 3);
            #pragma unroll
            for (int v = 0; v < 8; ++v) {
                if (nok) {
                    float x = acc[ti][tj][v] + bv;
                    if (ACT == 1) x = gelu_exact(x);
                    C[(size_t)(mbase + v) * ldC + n] = x;
                }
            }
        }
    }
}

// ---------------------------------------------------------------------------
// LayerNorm(1024) with optional residual, float4 vectorized, 1 block/row
// ---------------------------------------------------------------------------
__global__ __launch_bounds__(256)
void ln_kernel(const float* __restrict__ A, const float* __restrict__ R,
               const float* __restrict__ g, const float* __restrict__ b,
               float* __restrict__ out)
{
    __shared__ float s1[256], s2[256];
    const int row = blockIdx.x;
    const int tid = threadIdx.x;
    const int c0  = tid * 4;

    float4 va = *(const float4*)(A + (size_t)row * NE + c0);
    if (R) {
        float4 vr = *(const float4*)(R + (size_t)row * NE + c0);
        va.x += vr.x; va.y += vr.y; va.z += vr.z; va.w += vr.w;
    }
    float sum = va.x + va.y + va.z + va.w;
    float sq  = va.x * va.x + va.y * va.y + va.z * va.z + va.w * va.w;
    s1[tid] = sum; s2[tid] = sq;
    __syncthreads();
    for (int off = 128; off > 0; off >>= 1) {
        if (tid < off) { s1[tid] += s1[tid + off]; s2[tid] += s2[tid + off]; }
        __syncthreads();
    }
    float mean = s1[0] * (1.0f / NE);
    float var  = s2[0] * (1.0f / NE) - mean * mean;
    float inv  = rsqrtf(var + 1e-5f);
    float4 vg = *(const float4*)(g + c0);
    float4 vb = *(const float4*)(b + c0);
    float4 o;
    o.x = (va.x - mean) * inv * vg.x + vb.x;
    o.y = (va.y - mean) * inv * vg.y + vb.y;
    o.z = (va.z - mean) * inv * vg.z + vb.z;
    o.w = (va.w - mean) * inv * vg.w + vb.w;
    *(float4*)(out + (size_t)row * NE + c0) = o;
}

// ---------------------------------------------------------------------------
// Global sum + sumsq over a strided [M_ROWS, 20] view (Frobenius norm)
// ---------------------------------------------------------------------------
__global__ __launch_bounds__(256)
void norm_reduce_kernel(const float* __restrict__ p, int ld, float* __restrict__ out)
{
    __shared__ float s1[256], s2[256];
    const int tid = threadIdx.x;
    const int total = M_ROWS * HID;
    int base = blockIdx.x * 2048 + tid;
    float sum = 0.f, sq = 0.f;
    #pragma unroll
    for (int j = 0; j < 8; ++j) {
        int idx = base + j * 256;
        if (idx < total) {
            int r = idx / HID, c = idx % HID;
            float x = p[(size_t)r * ld + c];
            sum += x; sq += x * x;
        }
    }
    s1[tid] = sum; s2[tid] = sq;
    __syncthreads();
    for (int off = 128; off > 0; off >>= 1) {
        if (tid < off) { s1[tid] += s1[tid + off]; s2[tid] += s2[tid + off]; }
        __syncthreads();
    }
    if (tid == 0) { atomicAdd(&out[0], s1[0]); atomicAdd(&out[1], s2[0]); }
}

// ---------------------------------------------------------------------------
// Per-agent batch contraction: kvs[l,m,d], ksum[l,m], vsum[l,d]
// ---------------------------------------------------------------------------
__global__ __launch_bounds__(256)
void attn_reduce_kernel(const float* __restrict__ ks, const float* __restrict__ vs,
                        int ld, float* __restrict__ kvs, float* __restrict__ ksum,
                        float* __restrict__ vsum)
{
    const int l = blockIdx.x;
    const int t = threadIdx.x;
    #pragma unroll
    for (int pi = 0; pi < 2; ++pi) {
        int p = t + pi * 256;
        if (p < HID * HID) {
            int m = p / HID, d = p % HID;
            float a0 = 0.f, a1 = 0.f, a2 = 0.f;
            for (int bb = 0; bb < 2048; ++bb) {
                size_t base = ((size_t)bb * L_AG + l) * ld;
                float kv = ks[base + m];
                float vv = vs[base + d];
                a0 += kv * vv; a1 += kv; a2 += vv;
            }
            kvs[l * HID * HID + p] = a0;
            if (d == 0) ksum[l * HID + m] = a1;
            if (m == 0) vsum[l * HID + d] = a2;
        }
    }
}

// ---------------------------------------------------------------------------
// mid[b,l,d] = (s*qs.kvs + vsum) / (s*qs.ksum + B); s folds Frobenius norms
// scal: [0]=sum_q [1]=sumsq_q [2]=sum_k [3]=sumsq_k
// ---------------------------------------------------------------------------
__global__ __launch_bounds__(256)
void attn_combine_kernel(const float* __restrict__ qs, int ldq,
                         const float* __restrict__ kvs, const float* __restrict__ ksum,
                         const float* __restrict__ vsum, const float* __restrict__ scal,
                         float* __restrict__ mid)
{
    int idx = blockIdx.x * 256 + threadIdx.x;
    if (idx >= M_ROWS * HID) return;
    int r = idx / HID, d = idx % HID;
    int l = r & (L_AG - 1);

    bool cond = (scal[0] != 0.f) && (scal[2] != 0.f);
    float s = cond ? rsqrtf(scal[1]) * rsqrtf(scal[3]) : 1.0f;

    float q[HID];
    #pragma unroll
    for (int i = 0; i < HID; ++i) q[i] = qs[(size_t)r * ldq + i];

    float num = 0.f, den = 0.f;
    #pragma unroll
    for (int m = 0; m < HID; ++m) {
        num += q[m] * kvs[l * HID * HID + m * HID + d];
        den += q[m] * ksum[l * HID + m];
    }
    num = s * num + vsum[l * HID + d];
    den = s * den + 2048.0f;
    mid[(size_t)r * HID + d] = num / den;
}

// ---------------------------------------------------------------------------
// Host orchestration
// ---------------------------------------------------------------------------
extern "C" void kernel_launch(void* const* d_in, const int* in_sizes, int n_in,
                              void* d_out, int out_size, void* d_ws, size_t ws_size,
                              hipStream_t stream)
{
    (void)in_sizes; (void)n_in; (void)out_size; (void)ws_size;
    auto P = [&](int i) -> const float* { return (const float*)d_in[i]; };

    // input leaves (JAX pytree: dict keys sorted, lists in order)
    // 0 action, 1 obs_rep, 2 obs(unused); per block (26):
    // attn1{bk,bp,bq,bv,wk,wp,wq,wv}, attn2{...}, ln1_b,ln1_g,ln2_b,ln2_g,
    // ln3_b,ln3_g,mlp_b1,mlp_b2,mlp_w1,mlp_w2; then head_b1,head_b2,
    // head_ln_b,head_ln_g,head_w1,head_w2,ln_b,ln_g,w_act
    const float* action  = P(0);
    const float* obs_rep = P(1);
    const int HEAD = 3 + 8 * 26;  // 211

    // ---- fp32 workspace carve ----
    float* ws  = (float*)d_ws;
    size_t cur = 0;
    auto fall = [&](size_t n) { float* p = ws + cur; cur += (n + 3) & ~(size_t)3; return p; };
    float* x    = fall((size_t)M_ROWS * NE);
    float* t1   = fall((size_t)M_ROWS * NE);
    float* t2   = fall((size_t)M_ROWS * NE);
    float* kqv  = fall((size_t)M_ROWS * 60);
    float* mid  = fall((size_t)M_ROWS * HID);
    float* kvs  = fall(L_AG * HID * HID);
    float* ksum = fall(L_AG * HID);
    float* vsum = fall(L_AG * HID);
    float* scal = fall(4);
    float* b3   = fall(64);    // concat bias slot (60)
    float* bkv  = fall(48);    // concat bias slot (40)

    // ---- bf16 weight pool (16B-aligned allocations) ----
    unsigned short* pool = (unsigned short*)(ws + cur);
    size_t wcur = 0;
    auto balloc = [&](size_t n) { unsigned short* p = pool + wcur;
                                  wcur += (n + 7) & ~(size_t)7; return p; };
    auto conv = [&](const float* W, int K, int N, int nOff, unsigned short* dst) {
        dim3 g((N + 31) / 32, (K + 31) / 32);
        conv_t_kernel<<<g, 256, 0, stream>>>(W, dst, K, N, nOff);
    };

    struct BlkW { unsigned short *kqv1, *wp1, *kv2, *wq2, *wp2, *w1, *w2; };
    BlkW bw[8];
    for (int i = 0; i < 8; ++i) {
        int base = 3 + i * 26;
        bw[i].kqv1 = balloc((size_t)60 * NE);
        conv(P(base + 4),  NE, HID, 0,  bw[i].kqv1);   // wk -> rows 0..19
        conv(P(base + 6),  NE, HID, 20, bw[i].kqv1);   // wq -> rows 20..39
        conv(P(base + 7),  NE, HID, 40, bw[i].kqv1);   // wv -> rows 40..59
        bw[i].wp1 = balloc((size_t)NE * HID);
        conv(P(base + 5),  HID, NE, 0,  bw[i].wp1);    // wp [20,1024] -> [1024][20]
        bw[i].kv2 = balloc((size_t)40 * NE);
        conv(P(base + 12), NE, HID, 0,  bw[i].kv2);    // attn2 wk
        conv(P(base + 15), NE, HID, 20, bw[i].kv2);    // attn2 wv
        bw[i].wq2 = balloc((size_t)HID * NE);
        conv(P(base + 14), NE, HID, 0,  bw[i].wq2);    // attn2 wq
        bw[i].wp2 = balloc((size_t)NE * HID);
        conv(P(base + 13), HID, NE, 0,  bw[i].wp2);    // attn2 wp
        bw[i].w1 = balloc((size_t)NE * NE);
        conv(P(base + 24), NE, NE, 0,   bw[i].w1);     // mlp_w1
        bw[i].w2 = balloc((size_t)NE * NE);
        conv(P(base + 25), NE, NE, 0,   bw[i].w2);     // mlp_w2
    }
    unsigned short* wActT = balloc((size_t)NE * 65);
    conv(P(HEAD + 8), 65, NE, 0, wActT);
    unsigned short* h1T = balloc((size_t)NE * NE);
    conv(P(HEAD + 4), NE, NE, 0, h1T);
    unsigned short* h2T = balloc((size_t)64 * NE);
    conv(P(HEAD + 5), NE, 64, 0, h2T);

    // ---- launch helpers ----
    const dim3 blk(256);
    const int mblocks = M_ROWS / 128;   // 256
    auto gemm = [&](const float* A, const unsigned short* Wt, const float* bias,
                    float* C, int N, int K, int ldC, int act) {
        dim3 g((N + 63) / 64, mblocks);
        if (act) gemm_kernel<1><<<g, blk, 0, stream>>>(A, Wt, bias, C, M_ROWS, N, K, ldC);
        else     gemm_kernel<0><<<g, blk, 0, stream>>>(A, Wt, bias, C, M_ROWS, N, K, ldC);
    };
    const int nr_grid  = (M_ROWS * HID + 2047) / 2048;
    const int cmb_grid = (M_ROWS * HID + 255) / 256;

    auto attn_tail = [&](const float* ksp, const float* qsp, const float* vsp,
                         const unsigned short* wpT, const float* bp) {
        (void)hipMemsetAsync(scal, 0, 4 * sizeof(float), stream);
        norm_reduce_kernel<<<nr_grid, blk, 0, stream>>>(qsp, 60, scal + 0);
        norm_reduce_kernel<<<nr_grid, blk, 0, stream>>>(ksp, 60, scal + 2);
        attn_reduce_kernel<<<L_AG, blk, 0, stream>>>(ksp, vsp, 60, kvs, ksum, vsum);
        attn_combine_kernel<<<cmb_grid, blk, 0, stream>>>(qsp, 60, kvs, ksum, vsum,
                                                          scal, mid);
        gemm(mid, wpT, bp, t1, NE, HID, NE, 0);
    };

    // ---- embedding ----
    gemm(action, wActT, nullptr, t1, NE, 65, NE, 1);
    ln_kernel<<<M_ROWS, blk, 0, stream>>>(t1, nullptr, P(HEAD + 7), P(HEAD + 6), x);

    // ---- decoder blocks ----
    for (int i = 0; i < 8; ++i) {
        int base = 3 + i * 26;
        // self attention: fused QKV (cols: k=0..19, q=20..39, v=40..59)
        concat3_kernel<<<1, 64, 0, stream>>>(P(base + 0), P(base + 2), P(base + 3),
                                             b3, HID, HID, HID);
        gemm(x, bw[i].kqv1, b3, kqv, 60, NE, 60, 0);
        attn_tail(kqv + 0, kqv + 20, kqv + 40, bw[i].wp1, P(base + 1));
        ln_kernel<<<M_ROWS, blk, 0, stream>>>(t1, x, P(base + 17), P(base + 16), x);

        // cross attention: fused KV from x (cols 0..39), Q from obs_rep (cols 40..59)
        concat3_kernel<<<1, 64, 0, stream>>>(P(base + 8), P(base + 11), nullptr,
                                             bkv, HID, HID, 0);
        gemm(x, bw[i].kv2, bkv, kqv, 40, NE, 60, 0);
        gemm(obs_rep, bw[i].wq2, P(base + 10), kqv + 40, HID, NE, 60, 0);
        attn_tail(kqv + 0, kqv + 40, kqv + 20, bw[i].wp2, P(base + 9));
        ln_kernel<<<M_ROWS, blk, 0, stream>>>(t1, obs_rep, P(base + 19), P(base + 18), x);

        // MLP
        gemm(x,  bw[i].w1, P(base + 22), t1, NE, NE, NE, 1);
        gemm(t1, bw[i].w2, P(base + 23), t2, NE, NE, NE, 0);
        ln_kernel<<<M_ROWS, blk, 0, stream>>>(t2, x, P(base + 21), P(base + 20), x);
    }

    // ---- head ----
    gemm(x, h1T, P(HEAD + 0), t1, NE, NE, NE, 1);
    ln_kernel<<<M_ROWS, blk, 0, stream>>>(t1, nullptr, P(HEAD + 3), P(HEAD + 2), t1);
    gemm(t1, h2T, P(HEAD + 1), (float*)d_out, 64, NE, 64, 0);
}
